// MultiHeadVoting_64183991271582
// MI455X (gfx1250) — matrix operand order repaired and checked by reading, not compiled, not run on hardware
//
#include <hip/hip_runtime.h>

typedef __attribute__((ext_vector_type(16))) _Float16 v16h;
typedef __attribute__((ext_vector_type(8)))  float    v8f;

#define NHEAD 12
#define SEQ   785
#define PN    784      // patch_num
#define HDIM  28       // 28x28 image
#define TOPK  24

// Separable SAME conv kernel [1,2,1] as a 28x28 (padded to 32x32) symmetric
// tridiagonal matrix: conv2d(img, [1,2,1]^T[1,2,1], SAME) == T * img * T.
__device__ __forceinline__ _Float16 Tmat(int i, int k) {
  if (i >= HDIM || k >= HDIM) return (_Float16)0.f;
  int d = i - k; d = d < 0 ? -d : d;
  if (d == 0) return (_Float16)2.f;
  if (d == 1) return (_Float16)1.f;
  return (_Float16)0.f;
}

__global__ __launch_bounds__(256) void multihead_voting_kernel(
    const float* __restrict__ x, float* __restrict__ out) {
  __shared__ float    sc[PN];        // one head's score row
  __shared__ int      hist[PN];      // vote histogram
  __shared__ _Float16 img[32 * 32];  // padded count image (f16, exact ints)
  __shared__ float    tmpm[32 * 32]; // T * img
  __shared__ float    cv[32 * 32];   // (T * img) * T  == conv result
  __shared__ float    cb[PN];        // flattened conv counts

  const int b    = blockIdx.x;
  const int tid  = threadIdx.x;
  const int lane = tid & 31;
  const int lr   = lane & 15;  // lane % 16
  const int lh   = lane >> 4;  // lane half (0/1)

  for (int i = tid; i < PN; i += 256) hist[i] = 0;
  __syncthreads();

  // ---- Phase 1: per-head top-24 via stable rank counting, vote histogram ----
  for (int h = 0; h < NHEAD; ++h) {
    const float* row = x + (long long)(b * NHEAD + h) * SEQ * SEQ + 1; // x[b,h,0,1:]
    for (int i = tid; i < PN; i += 256) sc[i] = row[i];
    __syncthreads();
    for (int i = tid; i < PN; i += 256) {
      float v = sc[i];
      int rank = 0;
      for (int j = 0; j < PN; ++j) {
        float w = sc[j];
        rank += (w > v) || (w == v && j < i);  // stable descending (lax.top_k)
      }
      if (rank < TOPK) atomicAdd(&hist[i], 1);
    }
    __syncthreads();
  }

  // ---- Phase 2: pad histogram into 32x32 f16 image (values <= 12, exact) ----
  for (int i = tid; i < 32 * 32; i += 256) {
    int r = i >> 5, c = i & 31;
    img[i] = (r < HDIM && c < HDIM) ? (_Float16)(float)hist[r * HDIM + c]
                                    : (_Float16)0.f;
  }
  __syncthreads();

  // ---- Phase 3: conv = T * img * T via v_wmma_f32_16x16x32_f16 ----
  // All waves compute all 4 tiles redundantly (identical stores) so no
  // divergent branch surrounds WMMA: EXEC is all-ones as required.
  // A layout (16x32 f16): M = lane%16, K = 16*(lane/16) + e
  // B layout (32x16 f16): N = lane%16, K = 16*(lane/16) + e
  // D layout (16x16 f32): M = v + 8*(lane/16), N = lane%16
  for (int mt = 0; mt < 2; ++mt)
    for (int nt = 0; nt < 2; ++nt) {
      v16h a, bmat;
      for (int e = 0; e < 16; ++e) {
        int ka = (lh << 4) | e;
        a[e] = Tmat(mt * 16 + lr, ka);
        int kb = (lh << 4) | e;
        bmat[e] = img[kb * 32 + nt * 16 + lr];
      }
      v8f c = {};
      v8f d = __builtin_amdgcn_wmma_f32_16x16x32_f16(
          false, a, false, bmat, (short)0, c, false, false);
      for (int q = 0; q < 8; ++q)
        tmpm[(mt * 16 + (lh << 3) + q) * 32 + nt * 16 + lr] = d[q];
    }
  __syncthreads();

  for (int mt = 0; mt < 2; ++mt)
    for (int nt = 0; nt < 2; ++nt) {
      v16h a, bmat;
      for (int e = 0; e < 16; ++e) {
        int ka = (lh << 4) | e;
        a[e] = (_Float16)tmpm[(mt * 16 + lr) * 32 + ka]; // <=48, exact in f16
        int kb = (lh << 4) | e;
        bmat[e] = Tmat(nt * 16 + lr, kb);                // T symmetric
      }
      v8f c = {};
      v8f d = __builtin_amdgcn_wmma_f32_16x16x32_f16(
          false, a, false, bmat, (short)0, c, false, false);
      for (int q = 0; q < 8; ++q)
        cv[(mt * 16 + (lh << 3) + q) * 32 + nt * 16 + lr] = d[q];
    }
  __syncthreads();

  // ---- Phase 4: emit count output, then stable descending rank -> top-24 ----
  for (int i = tid; i < PN; i += 256) {
    float val = cv[(i / HDIM) * 32 + (i % HDIM)];
    cb[i] = val;
    out[32 * TOPK + (long long)b * PN + i] = val;   // count (B,784)
  }
  __syncthreads();

  for (int i = tid; i < PN; i += 256) {
    float v = cb[i];
    int rank = 0;
    for (int j = 0; j < PN; ++j) {
      float w = cb[j];
      rank += (w > v) || (w == v && j < i);  // matches jnp.argsort(-count) stable
    }
    if (rank < TOPK) out[b * TOPK + rank] = (float)(i + 1);  // patch_idx (B,24)
  }
}

extern "C" void kernel_launch(void* const* d_in, const int* in_sizes, int n_in,
                              void* d_out, int out_size, void* d_ws, size_t ws_size,
                              hipStream_t stream) {
  (void)in_sizes; (void)n_in; (void)out_size; (void)d_ws; (void)ws_size;
  const float* x = (const float*)d_in[0];
  // d_in[1] (3x3 kernel) is the fixed separable [1,2,1]x[1,2,1] — folded into Tmat.
  // d_in[2] (select_num) == 24 == TOPK.
  float* out = (float*)d_out;
  multihead_voting_kernel<<<32, 256, 0, stream>>>(x, out);
}
